// ModelA_GNN_58007828300461
// MI455X (gfx1250) — compile-verified, matrix operand order checked by reading, not run on hardware
//
#include <hip/hip_runtime.h>
#include <hip/hip_bf16.h>
#include <math.h>

typedef __attribute__((ext_vector_type(16))) _Float16 v16h;
typedef __attribute__((ext_vector_type(8)))  float    v8f;
typedef __attribute__((ext_vector_type(2)))  _Float16 half2v;

#define NN   350
#define NB   16
#define KNEI 10
#define BIGV 1e30f
#define PAIRS (NB*NN*NN)
#define TPW  4   // tiles (of 16 pairs) per wave in the edge kernel

// ---------------------------------------------------------------- utilities
__global__ void zero_kernel(float* p, int n) {
    int i = blockIdx.x * blockDim.x + threadIdx.x;
    if (i < n) p[i] = 0.0f;
}

// y = act(x @ w^T + b), torch layout w[OUT][IN]
__global__ void linear_rowmajor(const float* __restrict__ x, const float* __restrict__ w,
                                const float* __restrict__ bias, float* __restrict__ y,
                                int ROWS, int IN, int OUT, int relu) {
    int idx = blockIdx.x * blockDim.x + threadIdx.x;
    if (idx >= ROWS * OUT) return;
    int o = idx % OUT, r = idx / OUT;
    const float* xr = x + (long long)r * IN;
    const float* wr = w + (long long)o * IN;
    float acc = bias[o];
    for (int k = 0; k < IN; ++k) acc += xr[k] * wr[k];
    if (relu && acc < 0.f) acc = 0.f;
    y[idx] = acc;
}

// y = x @ w, weights stored [IN][OUT] (GCN layout), no bias/act
__global__ void linear_in_out(const float* __restrict__ x, const float* __restrict__ w,
                              float* __restrict__ y, int ROWS, int IN, int OUT) {
    int idx = blockIdx.x * blockDim.x + threadIdx.x;
    if (idx >= ROWS * OUT) return;
    int o = idx % OUT, r = idx / OUT;
    const float* xr = x + (long long)r * IN;
    float acc = 0.f;
    for (int k = 0; k < IN; ++k) acc += xr[k] * w[k * OUT + o];
    y[idx] = acc;
}

// ---------------------------------------------------------------- CNN blocks
template<int CIN, int COUT, int HIN>
__global__ void conv_relu_pool(const float* __restrict__ x, const float* __restrict__ w,
                               const float* __restrict__ bias, float* __restrict__ y) {
    const int HOUT = HIN / 2;
    int idx = blockIdx.x * blockDim.x + threadIdx.x;
    if (idx >= NB * COUT * HOUT * HOUT) return;
    int wo = idx % HOUT; int t = idx / HOUT;
    int ho = t % HOUT;   t /= HOUT;
    int oc = t % COUT;   int b = t / COUT;
    float m = -1e30f;
    #pragma unroll
    for (int dy = 0; dy < 2; ++dy)
    for (int dx = 0; dx < 2; ++dx) {
        int h = ho * 2 + dy, wc = wo * 2 + dx;
        float acc = bias[oc];
        for (int ic = 0; ic < CIN; ++ic) {
            const float* xp = x + (((long long)b * CIN + ic) * HIN) * HIN;
            const float* wp = w + ((long long)oc * CIN + ic) * 9;
            #pragma unroll
            for (int kh = 0; kh < 3; ++kh) {
                int hh = h + kh - 1;
                if (hh < 0 || hh >= HIN) continue;
                #pragma unroll
                for (int kw = 0; kw < 3; ++kw) {
                    int ww = wc + kw - 1;
                    if (ww < 0 || ww >= HIN) continue;
                    acc += xp[hh * HIN + ww] * wp[kh * 3 + kw];
                }
            }
        }
        acc = acc > 0.f ? acc : 0.f;
        m = acc > m ? acc : m;
    }
    y[idx] = m;
}

// ---------------------------------------------------------------- node count
__global__ void node_count_kernel(const float* __restrict__ h2, const float* __restrict__ w,
                                  const float* __restrict__ b, float* __restrict__ out) {
    int i = blockIdx.x * blockDim.x + threadIdx.x;
    if (i >= NB) return;
    float acc = b[0];
    for (int k = 0; k < 64; ++k) acc += h2[i * 64 + k] * w[k];
    out[i] = (1.0f / (1.0f + expf(-acc))) * (float)NN;
}

// ---------------------------------------------------------------- KNN build
__global__ void knn_kernel(const float* __restrict__ coords, const unsigned char* __restrict__ mask,
                           int* __restrict__ nbr, float* __restrict__ ew) {
    int idx = blockIdx.x * blockDim.x + threadIdx.x;
    if (idx >= NB * NN) return;
    int b = idx / NN, i = idx % NN;
    float xi = coords[idx * 2], yi = coords[idx * 2 + 1];
    bool mi = mask[idx] != 0;
    float d10[KNEI]; int i10[KNEI];
    #pragma unroll
    for (int k = 0; k < KNEI; ++k) { d10[k] = BIGV; i10[k] = 0; }
    for (int j = 0; j < NN; ++j) {
        float d;
        if (j == i || !mi || !mask[b * NN + j]) d = BIGV;
        else {
            float dx = xi - coords[(b * NN + j) * 2];
            float dy = yi - coords[(b * NN + j) * 2 + 1];
            d = sqrtf(fmaxf(dx * dx + dy * dy, 1e-12f));
        }
        if (d < d10[KNEI - 1]) {
            int k = KNEI - 1;
            while (k > 0 && d10[k - 1] > d) { d10[k] = d10[k - 1]; i10[k] = i10[k - 1]; --k; }
            d10[k] = d; i10[k] = j;
        }
    }
    float mf = mi ? 1.f : 0.f;
    #pragma unroll
    for (int k = 0; k < KNEI; ++k) {
        float dk = d10[k];
        ew[idx * KNEI + k] = expf(-0.5f * dk) * ((dk < BIGV * 0.5f) ? 1.f : 0.f) * mf;
        nbr[idx * KNEI + k] = i10[k];
    }
}

__global__ void scatter_kernel(const int* __restrict__ nbr, const float* __restrict__ ew,
                               float* __restrict__ M) {
    int idx = blockIdx.x * blockDim.x + threadIdx.x;
    if (idx >= NB * NN * KNEI) return;
    int b = idx / (NN * KNEI);
    int r = idx % (NN * KNEI);
    int i = r / KNEI;
    atomicAdd(&M[((long long)b * NN + nbr[idx]) * NN + i], ew[idx]);
}

__global__ void selfloop_kernel(float* __restrict__ M, const unsigned char* __restrict__ mask) {
    int idx = blockIdx.x * blockDim.x + threadIdx.x;
    if (idx >= NB * NN) return;
    int b = idx / NN, i = idx % NN;
    M[((long long)b * NN + i) * NN + i] += mask[idx] ? 1.0f : 0.0f;
}

__global__ void deg_kernel(const float* __restrict__ M, float* __restrict__ dinv) {
    int idx = blockIdx.x * blockDim.x + threadIdx.x;
    if (idx >= NB * NN) return;
    const float* row = M + (long long)idx * NN;
    float deg = 0.f;
    for (int s = 0; s < NN; ++s) deg += row[s];
    dinv[idx] = (deg > 0.f) ? rsqrtf(fmaxf(deg, 1e-12f)) : 0.f;
}

__global__ void norm_kernel(float* __restrict__ M, const float* __restrict__ dinv) {
    int idx = blockIdx.x * blockDim.x + threadIdx.x;
    if (idx >= PAIRS) return;
    int s = idx % NN;
    int t = (idx / NN) % NN;
    int b = idx / (NN * NN);
    M[idx] *= dinv[b * NN + t] * dinv[b * NN + s];
}

__global__ void spmm_kernel(const float* __restrict__ Mn, const float* __restrict__ X,
                            const float* __restrict__ bias, float* __restrict__ Y,
                            int D, const unsigned char* __restrict__ mask) {
    int idx = blockIdx.x * blockDim.x + threadIdx.x;
    if (idx >= NB * NN * D) return;
    int d = idx % D; int row = idx / D;
    int b = row / NN;
    const float* mr = Mn + (long long)row * NN;
    const float* xb = X + (long long)b * NN * D;
    float acc = bias[d];
    for (int s = 0; s < NN; ++s) acc += mr[s] * xb[s * D + d];
    acc = acc > 0.f ? acc : 0.f;
    if (mask) acc *= mask[row] ? 1.f : 0.f;
    Y[idx] = acc;
}

// a' = nf @ e1_w[:, :32]^T + e1_b (bias folded) ; c = nf @ e1_w[:, 32:]^T
// stored as f16 so the edge kernel builds he-fragments with packed f16 math
__global__ void ac_kernel(const float* __restrict__ nf, const float* __restrict__ e1_w,
                          const float* __restrict__ e1_b,
                          _Float16* __restrict__ Ah, _Float16* __restrict__ Ch) {
    int idx = blockIdx.x * blockDim.x + threadIdx.x;
    if (idx >= NB * NN * 64) return;
    int o = idx % 64; int row = idx / 64;
    const float* x = nf + (long long)row * 32;
    const float* wr = e1_w + (long long)o * 64;
    float aa = 0.f, cc = 0.f;
    for (int f = 0; f < 32; ++f) { aa += x[f] * wr[f]; cc += x[f] * wr[32 + f]; }
    Ah[idx] = (_Float16)(aa + e1_b[o]);
    Ch[idx] = (_Float16)cc;
}

// Precompute the 4 per-lane f16 fragments of e2_w used as the WMMA *A* operand
// (M = 32 out-features -> 2 M-tiles, K = 64 -> 2 K-tiles). 16-bit A 16x32
// layout per ISA 7.12.2: lane%16 -> row, element -> K (with +8 for lanes>=16).
__global__ void bfrag_kernel(const float* __restrict__ e2_w, v16h* __restrict__ Bf) {
    int lane = threadIdx.x;
    if (lane >= 32) return;
    int half_sel = lane >> 4, m0 = lane & 15;
    v16h f0, f1, f2, f3;
    #pragma unroll
    for (int e = 0; e < 16; ++e) {
        int v = e >> 1, pb = e & 1;
        int k = (v < 4 ? 2 * v : 16 + 2 * (v - 4)) + 8 * half_sel + pb;
        f0[e] = (_Float16)e2_w[m0 * 64 + k];              // mtile0, ktile0
        f1[e] = (_Float16)e2_w[(m0 + 16) * 64 + k];       // mtile1, ktile0
        f2[e] = (_Float16)e2_w[m0 * 64 + k + 32];         // mtile0, ktile1
        f3[e] = (_Float16)e2_w[(m0 + 16) * 64 + k + 32];  // mtile1, ktile1
    }
    Bf[0 * 32 + lane] = f0;
    Bf[1 * 32 + lane] = f1;
    Bf[2 * 32 + lane] = f2;
    Bf[3 * 32 + lane] = f3;
}

// ---------------------------------------------------------------- fused edge-pair WMMA
// D = e2_w x he^T : pairs ride the N (lane) dimension, out-features ride M.
// Each lane then holds 16 of the 32 out-features of ONE pair column, so the
// e3 dot is in-register except a single shfl_xor(16) per tile.
__global__ void edge_pair_kernel(const _Float16* __restrict__ Ah, const _Float16* __restrict__ Ch,
                                 const v16h* __restrict__ Bf,
                                 const float* __restrict__ e2_b, const float* __restrict__ e3_w,
                                 const float* __restrict__ e3_b,
                                 const unsigned char* __restrict__ mask,
                                 float* __restrict__ S, int numTiles) {
    int wave = (blockIdx.x * blockDim.x + threadIdx.x) >> 5;
    int lane = threadIdx.x & 31;
    int half_sel = lane >> 4;
    int n0 = lane & 15;

    // wave-invariant: e2_w A-fragments + per-lane epilogue constants
    v16h ae00 = Bf[0 * 32 + lane];   // features 0..15,  K 0..31
    v16h ae10 = Bf[1 * 32 + lane];   // features 16..31, K 0..31
    v16h ae01 = Bf[2 * 32 + lane];   // features 0..15,  K 32..63
    v16h ae11 = Bf[3 * 32 + lane];   // features 16..31, K 32..63
    float e2b_lo[8], e2b_hi[8], w3_lo[8], w3_hi[8];
    #pragma unroll
    for (int r = 0; r < 8; ++r) {
        int f = r + 8 * half_sel;
        e2b_lo[r] = e2_b[f];      w3_lo[r] = e3_w[f];
        e2b_hi[r] = e2_b[16 + f]; w3_hi[r] = e3_w[16 + f];
    }
    float b3 = e3_b[0];

    int tile0 = wave * TPW;
    for (int t = 0; t < TPW; ++t) {
        int tile = tile0 + t;
        if (tile >= numTiles) break;           // uniform per wave: EXEC stays all-1
        int tilebase = tile * 16;              // PAIRS % 16 == 0: no clamps needed

        // prefetch next tile's c-row for this lane
        int pn = tilebase + 16 + n0;
        if (pn < PAIRS) {
            int bn = pn / (NN * NN); int rn = pn % (NN * NN); int jn = rn % NN;
            __builtin_prefetch((const void*)(Ch + ((long long)bn * NN + jn) * 64), 0, 1);
        }

        int p = tilebase + n0;                 // this lane's pair column
        int b = p / (NN * NN);
        int rem = p % (NN * NN);
        int i = rem / NN, j = rem % NN;
        const _Float16* arow = Ah + ((long long)b * NN + i) * 64;
        const _Float16* crow = Ch + ((long long)b * NN + j) * 64;

        // he B-fragments: he = relu(a' + c), packed f16, pairs (k,k+1) per dword
        v16h hf0, hf1;
        #pragma unroll
        for (int v = 0; v < 8; ++v) {
            int k = (v < 4 ? 2 * v : 16 + 2 * (v - 4)) + 8 * half_sel;
            half2v r0 = *(const half2v*)(arow + k) + *(const half2v*)(crow + k);
            half2v r1 = *(const half2v*)(arow + k + 32) + *(const half2v*)(crow + k + 32);
            r0[0] = r0[0] > (_Float16)0 ? r0[0] : (_Float16)0;
            r0[1] = r0[1] > (_Float16)0 ? r0[1] : (_Float16)0;
            r1[0] = r1[0] > (_Float16)0 ? r1[0] : (_Float16)0;
            r1[1] = r1[1] > (_Float16)0 ? r1[1] : (_Float16)0;
            hf0[2 * v] = r0[0]; hf0[2 * v + 1] = r0[1];
            hf1[2 * v] = r1[0]; hf1[2 * v + 1] = r1[1];
        }

        v8f acc0 = {}; v8f acc1 = {};
        acc0 = __builtin_amdgcn_wmma_f32_16x16x32_f16(false, ae00, false, hf0, (short)0, acc0, false, false);
        acc0 = __builtin_amdgcn_wmma_f32_16x16x32_f16(false, ae01, false, hf1, (short)0, acc0, false, false);
        acc1 = __builtin_amdgcn_wmma_f32_16x16x32_f16(false, ae10, false, hf0, (short)0, acc1, false, false);
        acc1 = __builtin_amdgcn_wmma_f32_16x16x32_f16(false, ae11, false, hf1, (short)0, acc1, false, false);

        // epilogue: bias + relu + e3 dot, all in-lane; one cross-half combine
        float partial = 0.f;
        #pragma unroll
        for (int r = 0; r < 8; ++r) {
            float d0 = acc0[r] + e2b_lo[r]; d0 = d0 > 0.f ? d0 : 0.f;
            float d1 = acc1[r] + e2b_hi[r]; d1 = d1 > 0.f ? d1 : 0.f;
            partial += d0 * w3_lo[r] + d1 * w3_hi[r];
        }
        float tot = partial + __shfl_xor(partial, 16, 32);

        if (lane < 16) {                       // coalesced 16-wide store
            float sig = 1.0f / (1.0f + expf(-(tot + b3)));
            bool ok = (i != j) && mask[b * NN + i] && mask[b * NN + j];
            S[p] = ok ? sig : 0.0f;
        }
    }
}

__global__ void sym_kernel(const float* __restrict__ S, float* __restrict__ adj) {
    int idx = blockIdx.x * blockDim.x + threadIdx.x;
    if (idx >= PAIRS) return;
    int j = idx % NN;
    int i = (idx / NN) % NN;
    int b = idx / (NN * NN);
    adj[idx] = 0.5f * (S[idx] + S[((long long)b * NN + j) * NN + i]);
}

// ---------------------------------------------------------------- launcher
extern "C" void kernel_launch(void* const* d_in, const int* in_sizes, int n_in,
                              void* d_out, int out_size, void* d_ws, size_t ws_size,
                              hipStream_t stream) {
    const float* images  = (const float*)d_in[0];
    const unsigned char* masks = (const unsigned char*)d_in[1];
    const float* c1w = (const float*)d_in[2],  *c1b = (const float*)d_in[3];
    const float* c2w = (const float*)d_in[4],  *c2b = (const float*)d_in[5];
    const float* c3w = (const float*)d_in[6],  *c3b = (const float*)d_in[7];
    const float* fcw = (const float*)d_in[8],  *fcb = (const float*)d_in[9];
    const float* n1w = (const float*)d_in[10], *n1b = (const float*)d_in[11];
    const float* n2w = (const float*)d_in[12], *n2b = (const float*)d_in[13];
    const float* ncw = (const float*)d_in[14], *ncb = (const float*)d_in[15];
    const float* nnw = (const float*)d_in[16], *nnb = (const float*)d_in[17];
    const float* g1w = (const float*)d_in[18], *g1b = (const float*)d_in[19];
    const float* g2w = (const float*)d_in[20], *g2b = (const float*)d_in[21];
    const float* e1w = (const float*)d_in[22], *e1b = (const float*)d_in[23];
    const float* e2w = (const float*)d_in[24], *e2b = (const float*)d_in[25];
    const float* e3w = (const float*)d_in[26], *e3b = (const float*)d_in[27];

    float* out = (float*)d_out;
    float* coords_o = out;                                 // [16,350,2]
    float* ncnt_o   = out + 11200;                         // [16,1]
    float* adj_o    = out + 11216;                         // [16,350,350]
    float* nf_o     = out + 11216 + PAIRS;                 // [16,350,32]
    float* imf_o    = out + 11216 + PAIRS + NB * NN * 32;  // [16,256]

    float* ws = (float*)d_ws;
    float* conv1 = ws;                 ws += NB * 16 * 32 * 32;
    float* conv2 = ws;                 ws += NB * 32 * 16 * 16;
    float* conv3 = ws;                 ws += NB * 64 * 8 * 8;
    float* h1    = ws;                 ws += NB * 128;
    float* h2b   = ws;                 ws += NB * 64;
    float* ewb   = ws;                 ws += NB * NN * KNEI;
    int*   nbr   = (int*)ws;           ws += NB * NN * KNEI;
    float* M     = ws;                 ws += PAIRS;
    float* dinv  = ws;                 ws += NB * NN;
    float* Xg1   = ws;                 ws += NB * NN * 64;
    float* gh1   = ws;                 ws += NB * NN * 64;
    float* Xg2   = ws;                 ws += NB * NN * 32;
    _Float16* Ah = (_Float16*)ws;      ws += NB * NN * 32;   // NB*NN*64 halfs
    _Float16* Ch = (_Float16*)ws;      ws += NB * NN * 32;
    v16h* Bf     = (v16h*)ws;          ws += 1024;           // 4 frags x 32 lanes x 32B
    float* Sbuf  = ws;                 ws += PAIRS;

    const int T = 256;
    auto G = [](int n) { return (n + 255) / 256; };

    // CNN encoder
    conv_relu_pool<1, 16, 64><<<G(NB * 16 * 32 * 32), T, 0, stream>>>(images, c1w, c1b, conv1);
    conv_relu_pool<16, 32, 32><<<G(NB * 32 * 16 * 16), T, 0, stream>>>(conv1, c2w, c2b, conv2);
    conv_relu_pool<32, 64, 16><<<G(NB * 64 * 8 * 8), T, 0, stream>>>(conv2, c3w, c3b, conv3);
    linear_rowmajor<<<G(NB * 256), T, 0, stream>>>(conv3, fcw, fcb, imf_o, NB, 4096, 256, 1);

    // node regressor
    linear_rowmajor<<<G(NB * 128), T, 0, stream>>>(imf_o, n1w, n1b, h1, NB, 256, 128, 1);
    linear_rowmajor<<<G(NB * 64), T, 0, stream>>>(h1, n2w, n2b, h2b, NB, 128, 64, 1);
    linear_rowmajor<<<G(NB * 700), T, 0, stream>>>(h2b, ncw, ncb, coords_o, NB, 64, 700, 0);
    node_count_kernel<<<1, NB, 0, stream>>>(h2b, nnw, nnb, ncnt_o);

    // KNN graph + gcn_norm
    knn_kernel<<<G(NB * NN), T, 0, stream>>>(coords_o, masks, nbr, ewb);
    zero_kernel<<<G(PAIRS), T, 0, stream>>>(M, PAIRS);
    scatter_kernel<<<G(NB * NN * KNEI), T, 0, stream>>>(nbr, ewb, M);
    selfloop_kernel<<<G(NB * NN), T, 0, stream>>>(M, masks);
    deg_kernel<<<G(NB * NN), T, 0, stream>>>(M, dinv);
    norm_kernel<<<G(PAIRS), T, 0, stream>>>(M, dinv);

    // GCN
    linear_in_out<<<G(NB * NN * 64), T, 0, stream>>>(coords_o, g1w, Xg1, NB * NN, 2, 64);
    spmm_kernel<<<G(NB * NN * 64), T, 0, stream>>>(M, Xg1, g1b, gh1, 64, nullptr);
    linear_in_out<<<G(NB * NN * 32), T, 0, stream>>>(gh1, g2w, Xg2, NB * NN, 64, 32);
    spmm_kernel<<<G(NB * NN * 32), T, 0, stream>>>(M, Xg2, g2b, nf_o, 32, masks);

    // edge predictor: split e1 GEMM (f16 outputs, bias folded), f16 e2 fragments,
    // then fused WMMA pair kernel (pairs on N, features on M)
    ac_kernel<<<G(NB * NN * 64), T, 0, stream>>>(nf_o, e1w, e1b, Ah, Ch);
    bfrag_kernel<<<1, 32, 0, stream>>>(e2w, Bf);
    int numTiles = PAIRS / 16;                        // 122500
    int numWaves = (numTiles + TPW - 1) / TPW;        // 30625
    int waveBlocks = (numWaves + 7) / 8;              // 8 waves / 256-thread block
    edge_pair_kernel<<<waveBlocks, 256, 0, stream>>>(Ah, Ch, Bf, e2b, e3w, e3b,
                                                     masks, Sbuf, numTiles);
    sym_kernel<<<G(PAIRS), T, 0, stream>>>(Sbuf, adj_o);
}
